// Attention_36240934044205
// MI455X (gfx1250) — matrix operand order repaired and checked by reading, not compiled, not run on hardware
//
#include <hip/hip_runtime.h>

// ---------------- types matching gfx1250 WMMA / TDM builtins ----------------
typedef __attribute__((ext_vector_type(16))) __bf16 v16bf;
typedef __attribute__((ext_vector_type(8)))  __bf16 v8bf;
typedef __attribute__((ext_vector_type(4)))  __bf16 v4bf;
typedef __attribute__((ext_vector_type(8)))  float  v8f;
typedef __attribute__((ext_vector_type(4)))  unsigned int v4u;
typedef __attribute__((ext_vector_type(8)))  int v8i;
typedef __attribute__((ext_vector_type(4)))  int v4i;

// ---------------- problem constants ----------------
#define B_    8
#define TQ    2048
#define TK    2048
#define DD    1024
#define BM    16        // query rows per block
#define BN    32        // key rows per k-step
#define NW    4         // waves per block (wave32)
#define DQ    256       // D columns owned by each wave (DD / NW)
#define NTILE 16        // 16x16 output tiles per wave (DQ/16)

// LDS strides (elements), padded to stagger banks
#define QS_S  1032
#define KS_S  1032
#define KT_S  40
#define PS_S  40
#define SR_S  36

// LDS byte offsets (keep everything 16B aligned)
#define OFF_QS 0
#define OFF_KS (OFF_QS + BM * QS_S * 2)          //  33024
#define OFF_KT (OFF_KS + BN * KS_S * 2)          //  99072
#define OFF_PS (OFF_KT + DD * KT_S * 2)          // 180992
#define OFF_SR (OFF_PS + BM * PS_S * 2)          // 182272
#define OFF_MS (OFF_SR + NW * BM * SR_S * 4)     // 191488
#define OFF_LS (OFF_MS + BM * 4)
#define OFF_CR (OFF_LS + BM * 4)
#define OFF_STG (OFF_CR + BM * 4)                // 191680 (16B aligned)
#define SMEM_BYTES (OFF_STG + BN * DD * 4)       // 322752 B  (<= 320 KB/WGP)

// ---------------- TDM availability ----------------
#if defined(__has_builtin)
#if __has_builtin(__builtin_amdgcn_tensor_load_to_lds) && \
    __has_builtin(__builtin_amdgcn_s_wait_tensorcnt)
#define USE_TDM 1
#endif
#endif
#ifndef USE_TDM
#define USE_TDM 0
#endif

// ---------------- helpers ----------------
static __device__ __forceinline__ v16bf ldfrag(const __bf16* p0, const __bf16* p1) {
    v8bf lo = *(const v8bf*)p0;
    v8bf hi = *(const v8bf*)p1;
    return __builtin_shufflevector(lo, hi, 0,1,2,3,4,5,6,7,8,9,10,11,12,13,14,15);
}

static __device__ __forceinline__ float wave_max(float v) {
    #pragma unroll
    for (int m = 16; m >= 1; m >>= 1) v = fmaxf(v, __shfl_xor(v, m, 32));
    return v;
}
static __device__ __forceinline__ float wave_sum(float v) {
    #pragma unroll
    for (int m = 16; m >= 1; m >>= 1) v += __shfl_xor(v, m, 32);
    return v;
}

static __device__ __forceinline__ v8f wmma_bf16(v16bf a, v16bf b, v8f c) {
    return __builtin_amdgcn_wmma_f32_16x16x32_bf16(false, a, false, b, (short)0, c,
                                                   false, false);
}

#if USE_TDM
// Async DMA of a [rows x cols] f32 tile (row stride = cols) from global into LDS.
// D# fields per CDNA5 ISA sec.8: group0 = {count, lds_addr,
// global_addr[56:0], type=2}; group1 = {wg_mask=0, data_size=4B, tensor_dim0/1,
// tile_dim0/1, tensor_dim0_stride}. 2-D tile: groups 2/3 zero-filled.
static __device__ __forceinline__ void tdm_load_f32_tile(const float* gsrc,
                                                         unsigned lds_off,
                                                         int rows, int cols) {
    unsigned long long ga = (unsigned long long)(size_t)gsrc;
    v4u g0;
    g0[0] = 1u;                                         // count=1 (valid user D#)
    g0[1] = lds_off;                                    // lds_addr (bytes)
    g0[2] = (unsigned)(ga & 0xffffffffu);               // global_addr[31:0]
    g0[3] = (unsigned)((ga >> 32) & 0x01ffffffu)        // global_addr[56:32]
          | 0x80000000u;                                // type = 2 ("image")
    v8i g1;
    g1[0] = (int)(2u << 16);                            // data_size = 2 -> 4 bytes
    g1[1] = (int)((unsigned)(cols & 0xffff) << 16);     // tensor_dim0[15:0]
    g1[2] = (int)(((unsigned)cols >> 16)                // tensor_dim0[31:16]
          | ((unsigned)(rows & 0xffff) << 16));         // tensor_dim1[15:0]
    g1[3] = (int)((unsigned)(cols & 0xffff) << 16);     // tile_dim0 (tdim1 hi = 0)
    g1[4] = rows;                                       // tile_dim1 (tile_dim2 = 0)
    g1[5] = cols;                                       // tensor_dim0_stride[31:0]
    g1[6] = 0;                                          // stride hi / dim1_stride
    g1[7] = 0;
    v4i g2 = {0, 0, 0, 0};
    v4i g3 = {0, 0, 0, 0};
#if __clang_major__ >= 23
    v8i g4 = {0, 0, 0, 0, 0, 0, 0, 0};
    __builtin_amdgcn_tensor_load_to_lds(g0, g1, g2, g3, g4, 0);
#else
    __builtin_amdgcn_tensor_load_to_lds(g0, g1, g2, g3, 0);
#endif
}
#endif

// ---------------- fused flash-attention kernel ----------------
__global__ __launch_bounds__(128)
void Attention_36240934044205_kernel(const float* __restrict__ qg_,
                                     const float* __restrict__ kg_,
                                     float* __restrict__ out) {
    extern __shared__ __attribute__((aligned(16))) char smem[];
    __bf16* Qs = (__bf16*)(smem + OFF_QS);   // [BM][QS_S]  Q tile (bf16)
    __bf16* Ks = (__bf16*)(smem + OFF_KS);   // [BN][KS_S]  K tile, natural
    __bf16* Kt = (__bf16*)(smem + OFF_KT);   // [DD][KT_S]  K tile, transposed
    __bf16* Ps = (__bf16*)(smem + OFF_PS);   // [BM][PS_S]  exp(S-m) tile (bf16)
    float*  Sr = (float*)(smem + OFF_SR);    // [NW][BM][SR_S] partial S
    float*  Ms = (float*)(smem + OFF_MS);    // [BM] running max
    float*  Ls = (float*)(smem + OFF_LS);    // [BM] running sum
    float*  Cr = (float*)(smem + OFF_CR);    // [BM] rescale factor this step
    float*  Stg = (float*)(smem + OFF_STG);  // [BN][DD] raw f32 K staging (TDM)

    const int tid  = threadIdx.x;
    const int lane = tid & 31;
    const int wid  = tid >> 5;
    const int bidx = blockIdx.x;
    const int b    = bidx >> 7;        // / (TQ/BM)
    const int q0   = (bidx & 127) * BM;

    // ---- load + convert Q tile (f32 -> bf16) ----
    const float4* qg = (const float4*)(qg_ + ((size_t)b * TQ + q0) * DD);
    for (int i = tid; i < BM * DD / 4; i += 128) {
        float4 v = qg[i];
        int row = i >> 8;
        int c4  = (i & 255) * 4;
        v4bf pk = {(__bf16)v.x, (__bf16)v.y, (__bf16)v.z, (__bf16)v.w};
        *(v4bf*)(Qs + row * QS_S + c4) = pk;
    }
    if (tid < BM) { Ms[tid] = -3.0e38f; Ls[tid] = 0.0f; Cr[tid] = 1.0f; }

    v8f acc[NTILE];
    v8f zacc = {};
    #pragma unroll
    for (int t = 0; t < NTILE; ++t) acc[t] = zacc;

    const int rlo    = lane & 15;
    const int hi16   = (lane & 16) ? 1 : 0;
    const int koff_a = hi16 ? 16 : 0;
    const int koff_b = hi16 ? 16 : 0;
    const int rowoff = hi16 ? 8 : 0;
    const int dq     = wid * DQ;

    const float4* kg = (const float4*)(kg_ + (size_t)b * TK * DD);
    (void)kg;

#if USE_TDM
    const unsigned stg_off = (unsigned)(size_t)(void*)Stg;  // LDS byte offset
    if (wid == 0)   // one TDM op per workgroup: prefetch K tile 0
        tdm_load_f32_tile(kg_ + (size_t)b * TK * DD, stg_off, BN, DD);
#endif

    for (int kt = 0; kt < TK / BN; ++kt) {
#if USE_TDM
        if (wid == 0) __builtin_amdgcn_s_wait_tensorcnt(0);
        __syncthreads();   // staging ready; prior step done reading Ks/Kt/Ps

        // ---- convert staged f32 tile -> bf16 in both layouts (LDS->LDS) ----
        const float4* sg = (const float4*)Stg;
        for (int i = tid; i < BN * DD / 4; i += 128) {
            float4 v = sg[i];
            int row = i >> 8;
            int c4  = (i & 255) * 4;
            __bf16 e0 = (__bf16)v.x, e1 = (__bf16)v.y;
            __bf16 e2 = (__bf16)v.z, e3 = (__bf16)v.w;
            v4bf pk = {e0, e1, e2, e3};
            *(v4bf*)(Ks + row * KS_S + c4) = pk;
            Kt[(c4 + 0) * KT_S + row] = e0;
            Kt[(c4 + 1) * KT_S + row] = e1;
            Kt[(c4 + 2) * KT_S + row] = e2;
            Kt[(c4 + 3) * KT_S + row] = e3;
        }
        __syncthreads();   // Ks/Kt published; staging free for next DMA

        if (wid == 0 && kt + 1 < TK / BN)   // async prefetch of next K tile
            tdm_load_f32_tile(kg_ + ((size_t)b * TK + (size_t)(kt + 1) * BN) * DD,
                              stg_off, BN, DD);
#else
        const int kn0 = kt * BN;
        __syncthreads();
        for (int i = tid; i < BN * DD / 4; i += 128) {
            float4 v = kg[(size_t)kn0 * (DD / 4) + i];
            int row = i >> 8;
            int c4  = (i & 255) * 4;
            __bf16 e0 = (__bf16)v.x, e1 = (__bf16)v.y;
            __bf16 e2 = (__bf16)v.z, e3 = (__bf16)v.w;
            v4bf pk = {e0, e1, e2, e3};
            *(v4bf*)(Ks + row * KS_S + c4) = pk;
            Kt[(c4 + 0) * KT_S + row] = e0;
            Kt[(c4 + 1) * KT_S + row] = e1;
            Kt[(c4 + 2) * KT_S + row] = e2;
            Kt[(c4 + 3) * KT_S + row] = e3;
        }
        __syncthreads();
#endif

        // ---- phase 1: partial S = Q . K^T over this wave's D-quarter ----
        v8f cs0 = zacc, cs1 = zacc;
        #pragma unroll
        for (int kk = 0; kk < DQ; kk += 32) {
            const __bf16* ap = Qs + rlo * QS_S + dq + kk + koff_a;
            v16bf a = ldfrag(ap, ap + 8);
            const __bf16* bp0 = Ks + rlo * KS_S + dq + kk + koff_b;
            cs0 = wmma_bf16(a, ldfrag(bp0, bp0 + 8), cs0);
            const __bf16* bp1 = Ks + (16 + rlo) * KS_S + dq + kk + koff_b;
            cs1 = wmma_bf16(a, ldfrag(bp1, bp1 + 8), cs1);
        }
        float* srw = Sr + wid * BM * SR_S;
        #pragma unroll
        for (int r = 0; r < 8; ++r) {
            srw[(r + rowoff) * SR_S + rlo]      = cs0[r];
            srw[(r + rowoff) * SR_S + 16 + rlo] = cs1[r];
        }
        __syncthreads();

        // ---- online softmax: wave handles 4 rows, lane == column ----
        #pragma unroll
        for (int rr = 0; rr < 4; ++rr) {
            int r = wid * 4 + rr;
            float s = Sr[0 * BM * SR_S + r * SR_S + lane]
                    + Sr[1 * BM * SR_S + r * SR_S + lane]
                    + Sr[2 * BM * SR_S + r * SR_S + lane]
                    + Sr[3 * BM * SR_S + r * SR_S + lane];
            float mt = wave_max(s);
            float mo = Ms[r];
            float mn = fmaxf(mo, mt);
            float p  = __expf(s - mn);
            float st = wave_sum(p);
            float cf = __expf(mo - mn);
            if (lane == 0) { Ms[r] = mn; Ls[r] = Ls[r] * cf + st; Cr[r] = cf; }
            Ps[r * PS_S + lane] = (__bf16)p;
        }
        __syncthreads();

        // ---- rescale accumulator, then phase 2: O += P . K (D-quarter) ----
        float cf8[8];
        #pragma unroll
        for (int r = 0; r < 8; ++r) cf8[r] = Cr[r + rowoff];
        #pragma unroll
        for (int t = 0; t < NTILE; ++t)
            #pragma unroll
            for (int r = 0; r < 8; ++r) acc[t][r] *= cf8[r];

        const __bf16* pap = Ps + rlo * PS_S + koff_a;
        v16bf pa = ldfrag(pap, pap + 8);
        #pragma unroll
        for (int t = 0; t < NTILE; ++t) {
            const __bf16* bp = Kt + (dq + t * 16 + rlo) * KT_S + koff_b;
            acc[t] = wmma_bf16(pa, ldfrag(bp, bp + 8), acc[t]);
        }
    }

    // ---- epilogue: normalize by running sum and store f32 output ----
    float linv[8];
    #pragma unroll
    for (int r = 0; r < 8; ++r) linv[r] = 1.0f / Ls[r + rowoff];
    #pragma unroll
    for (int t = 0; t < NTILE; ++t) {
        int n = dq + t * 16 + rlo;
        #pragma unroll
        for (int r = 0; r < 8; ++r) {
            int m = r + rowoff;
            out[((size_t)b * TQ + q0 + m) * DD + n] = acc[t][r] * linv[r];
        }
    }
}

extern "C" void kernel_launch(void* const* d_in, const int* in_sizes, int n_in,
                              void* d_out, int out_size, void* d_ws, size_t ws_size,
                              hipStream_t stream) {
    (void)in_sizes; (void)n_in; (void)out_size; (void)d_ws; (void)ws_size;
    const float* q = (const float*)d_in[0];
    const float* k = (const float*)d_in[1];
    float* out = (float*)d_out;
    dim3 grid(B_ * (TQ / BM));   // 1024 blocks: one per (batch, 16-row q tile)
    dim3 block(128);             // 4 wave32 waves; ~315 KB LDS -> 1 WG per WGP
    Attention_36240934044205_kernel<<<grid, block, SMEM_BYTES, stream>>>(q, k, out);
}
